// SliceAttention_41515153883184
// MI455X (gfx1250) — matrix operand order repaired and checked by reading, not compiled, not run on hardware
//
#include <hip/hip_runtime.h>
#include <hip/hip_bf16.h>

typedef __attribute__((ext_vector_type(16))) __bf16 v16bf;
typedef __attribute__((ext_vector_type(8)))  __bf16 v8bf;
typedef __attribute__((ext_vector_type(8)))  float  v8f;

#define B_  4
#define N_  16384
#define HD_ 1024
#define H_  8
#define M_  32
#define D_  128
#define BN_TOT 65536L  /* B_*N_ */

#if __has_builtin(__builtin_amdgcn_global_load_async_to_lds_b128)
#define ASYNC_LDS 1
typedef int v4i_vs __attribute__((vector_size(16)));
typedef __attribute__((address_space(1))) v4i_vs* as1_v4i;
typedef __attribute__((address_space(3))) v4i_vs* as3_v4i;
#else
#define ASYNC_LDS 0
#endif

__device__ __forceinline__ void async_wait_all() {
#if __has_builtin(__builtin_amdgcn_s_wait_asynccnt)
  __builtin_amdgcn_s_wait_asynccnt(0);
#elif ASYNC_LDS
  asm volatile("s_wait_asynccnt 0" ::: "memory");
#endif
}

// ---------------------------------------------------------------- utilities
__global__ void k_f32_to_bf16(const float* __restrict__ s, __bf16* __restrict__ d, long n) {
  long i = (long)blockIdx.x * blockDim.x + threadIdx.x;
  long stride = (long)gridDim.x * blockDim.x;
  for (; i < n; i += stride) d[i] = (__bf16)s[i];
}

__global__ void k_zero_f32(float* __restrict__ p, long n) {
  long i = (long)blockIdx.x * blockDim.x + threadIdx.x;
  long stride = (long)gridDim.x * blockDim.x;
  for (; i < n; i += stride) p[i] = 0.f;
}

__global__ void k_misc(const float* __restrict__ wtq_bias, float* __restrict__ dout) {
  int t = threadIdx.x;
  if (t == 0) dout[83886080] = 1.0f;                  // temperature
  if (t < H_ * M_) dout[83886081 + t] = wtq_bias[t];  // bias [H,M,1]
}

// ---------------------------------------------------------------- WMMA GEMM
// C[m,n] = sum_k A[m,k] * Bw[n,k] + bias[n]
// A: [Mrows x K] bf16 row-major, Bw: [Ncols x K] bf16 row-major (== W as stored)
#define BM 128
#define BNT 128
#define BK 32

template <bool OUT_BF16>
__global__ __launch_bounds__(256) void gemm_bf16(
    const __bf16* __restrict__ A, const __bf16* __restrict__ Bw,
    const float* __restrict__ bias, void* __restrict__ Cout,
    int Mrows, int Ncols, int K) {
  __shared__ __bf16 As[2][BM * BK];
  __shared__ __bf16 Bs[2][BNT * BK];
  const int tid  = threadIdx.x;
  const int wave = tid >> 5;
  const int lane = tid & 31;
  const int wm   = wave & 3;   // 4 waves along M
  const int wn   = wave >> 2;  // 2 waves along N
  const int m0   = blockIdx.y * BM;
  const int n0   = blockIdx.x * BNT;
  const int hf   = lane >> 4;
  const int r16  = lane & 15;

  v8f acc[2][4];
#pragma unroll
  for (int i = 0; i < 2; ++i)
#pragma unroll
    for (int j = 0; j < 4; ++j)
#pragma unroll
      for (int e = 0; e < 8; ++e) acc[i][j][e] = 0.f;

  // each thread moves 2x16B of A and 2x16B of B per tile
  auto issue_tile = [&](int buf, int k0) {
#pragma unroll
    for (int p = 0; p < 2; ++p) {
      int c   = tid + p * 256;   // 0..511
      int row = c >> 2;          // 0..127
      int seg = (c & 3) << 3;    // 0,8,16,24
      const __bf16* ga = A  + (size_t)(m0 + row) * K + k0 + seg;
      const __bf16* gb = Bw + (size_t)(n0 + row) * K + k0 + seg;
      __bf16* la = &As[buf][row * BK + seg];
      __bf16* lb = &Bs[buf][row * BK + seg];
#if ASYNC_LDS
      // CDNA5 async LDS-DMA path: global -> LDS without VGPR round trip (ASYNCcnt)
      __builtin_amdgcn_global_load_async_to_lds_b128(
          (as1_v4i)(void*)ga, (as3_v4i)(void*)la, 0, 0);
      __builtin_amdgcn_global_load_async_to_lds_b128(
          (as1_v4i)(void*)gb, (as3_v4i)(void*)lb, 0, 0);
#else
      *(v8bf*)la = *(const v8bf*)ga;
      *(v8bf*)lb = *(const v8bf*)gb;
#endif
    }
  };

  int cur = 0;
  issue_tile(0, 0);
  async_wait_all();
  __syncthreads();

  for (int k0 = 0; k0 < K; k0 += BK) {
    const int  nxt      = cur ^ 1;
    const bool has_next = (k0 + BK < K);
    if (has_next) issue_tile(nxt, k0 + BK);  // DMA overlaps this step's compute

    v16bf af[2], bfr[4];
#pragma unroll
    for (int i = 0; i < 2; ++i) {
      // A 16x32 bf16 fragment: lanes 0-15 hold K=0..7 / 16..23, lanes 16-31 K=8..15 / 24..31
      const __bf16* p = As[cur] + (wm * 32 + i * 16 + r16) * BK + hf * 8;
      v8bf lo = *(const v8bf*)p;
      v8bf hi = *(const v8bf*)(p + 16);
      v16bf f;
#pragma unroll
      for (int j = 0; j < 8; ++j) { f[j] = lo[j]; f[8 + j] = hi[j]; }
      af[i] = f;
    }
#pragma unroll
    for (int j = 0; j < 4; ++j) {
      // B 32x16 bf16 fragment: lanes 0-15 hold K=0..7 / 16..23, lanes 16-31 K=8..15 / 24..31
      const __bf16* p = Bs[cur] + (wn * 64 + j * 16 + r16) * BK + hf * 8;
      v8bf lo = *(const v8bf*)p;
      v8bf hi = *(const v8bf*)(p + 16);
      v16bf f;
#pragma unroll
      for (int q = 0; q < 8; ++q) { f[q] = lo[q]; f[8 + q] = hi[q]; }
      bfr[j] = f;
    }
#pragma unroll
    for (int i = 0; i < 2; ++i)
#pragma unroll
      for (int j = 0; j < 4; ++j)
        acc[i][j] = __builtin_amdgcn_wmma_f32_16x16x32_bf16(
            false, af[i], false, bfr[j], (short)0, acc[i][j], false, false);

    if (has_next) async_wait_all();  // this step's DMA had the whole compute to land
    __syncthreads();                 // publish nxt tile, retire cur tile
    cur = nxt;
  }

#pragma unroll
  for (int i = 0; i < 2; ++i)
#pragma unroll
    for (int j = 0; j < 4; ++j) {
      int gn = n0 + wn * 64 + j * 16 + r16;
      float bv = bias[gn];
#pragma unroll
      for (int r2 = 0; r2 < 8; ++r2) {
        int gm = m0 + wm * 32 + i * 16 + hf * 8 + r2;
        float v = acc[i][j][r2] + bv;
        if (OUT_BF16)
          ((__bf16*)Cout)[(size_t)gm * Ncols + gn] = (__bf16)v;
        else
          ((float*)Cout)[(size_t)gm * Ncols + gn] = v;
      }
    }
}

// ------------------------------------------------- scores + softmax + slice_token
__global__ __launch_bounds__(256) void k_scores(
    const __bf16* __restrict__ kvb, const float* __restrict__ wtq,
    float* __restrict__ swp, float* __restrict__ stok, float* __restrict__ snorm) {
  __shared__ float  wq[M_ * 130];        // padded to avoid bank conflicts
  __shared__ __bf16 xt[128 * D_];        // xk then xv tile
  __shared__ float  wbuf[128 * M_];
  const int tid = threadIdx.x;
  const int wave = tid >> 5, lane = tid & 31;
  const long tg = (long)blockIdx.x * 128;
  const int b = (int)(tg / N_);
  const int n0 = (int)(tg % N_);

  for (int h = 0; h < H_; ++h) {
    for (int idx = tid; idx < M_ * D_; idx += 256) {
      int m = idx >> 7, dd = idx & 127;
      wq[m * 130 + dd] = wtq[((size_t)h * M_ + m) * D_ + dd];
    }
    {
      const __bf16* src = kvb + ((size_t)(b * N_ + n0)) * (2 * HD_) + h * D_;
#pragma unroll
      for (int p = 0; p < 8; ++p) {
        int c = tid + p * 256;
        int tok = c >> 4, seg = (c & 15) * 8;
        *(v8bf*)(xt + tok * D_ + seg) = *(const v8bf*)(src + (size_t)tok * (2 * HD_) + seg);
      }
    }
    __syncthreads();
    // each wave: one token at a time, lane == m; softmax over the 32 lanes (slice axis)
    for (int t2 = wave; t2 < 128; t2 += 8) {
      int m = lane;
      float s = 0.f;
      for (int dd = 0; dd < D_; ++dd) s += wq[m * 130 + dd] * (float)xt[t2 * D_ + dd];
      float mx = s;
      for (int o = 16; o > 0; o >>= 1) mx = fmaxf(mx, __shfl_xor(mx, o, 32));
      float e = __expf(s - mx);
      float sm = e;
      for (int o = 16; o > 0; o >>= 1) sm += __shfl_xor(sm, o, 32);
      float w = e / sm;
      wbuf[t2 * M_ + m] = w;
      swp[((size_t)((b * H_ + h) * M_ + m)) * N_ + n0 + t2] = w;
    }
    __syncthreads();
    {
      const __bf16* src = kvb + ((size_t)(b * N_ + n0)) * (2 * HD_) + HD_ + h * D_;
#pragma unroll
      for (int p = 0; p < 8; ++p) {
        int c = tid + p * 256;
        int tok = c >> 4, seg = (c & 15) * 8;
        *(v8bf*)(xt + tok * D_ + seg) = *(const v8bf*)(src + (size_t)tok * (2 * HD_) + seg);
      }
    }
    __syncthreads();
    {
      int m = tid >> 3;
      int dseg = (tid & 7) * 16;
      float accv[16];
#pragma unroll
      for (int j = 0; j < 16; ++j) accv[j] = 0.f;
      float nrm = 0.f;
      for (int tok = 0; tok < 128; ++tok) {
        float wv = wbuf[tok * M_ + m];
        nrm += wv;
#pragma unroll
        for (int j = 0; j < 16; ++j) accv[j] += wv * (float)xt[tok * D_ + dseg + j];
      }
      float* dst = stok + ((size_t)((b * H_ + h) * M_ + m)) * D_ + dseg;
#pragma unroll
      for (int j = 0; j < 16; ++j) atomicAdd(dst + j, accv[j]);
      if ((tid & 7) == 0) atomicAdd(snorm + (b * H_ + h) * M_ + m, nrm);
    }
    __syncthreads();
  }
}

// ------------------------------------------------- tiny per-(b,h) token attention
__global__ __launch_bounds__(256) void k_token_attn(
    const float* __restrict__ stok, const float* __restrict__ snorm,
    const float* __restrict__ qkvw, float* __restrict__ attn_out,
    float* __restrict__ otok) {
  __shared__ float st[M_ * D_];
  __shared__ float qk[M_ * 384];
  __shared__ float dm[M_ * M_];
  __shared__ float aw[M_ * M_];
  const int tid = threadIdx.x;
  const int b = blockIdx.x >> 3, h = blockIdx.x & 7;
  const int bh = b * H_ + h;

  for (int idx = tid; idx < M_ * D_; idx += 256) {
    int m = idx >> 7;
    st[idx] = stok[(size_t)bh * M_ * D_ + idx] / (snorm[bh * M_ + m] + 1e-5f);
  }
  __syncthreads();
  const float* W = qkvw + (size_t)h * D_ * 384;
  for (int idx = tid; idx < M_ * 384; idx += 256) {
    int m = idx / 384, e = idx % 384;
    float s = 0.f;
    for (int dd = 0; dd < D_; ++dd) s += st[m * D_ + dd] * W[dd * 384 + e];
    qk[idx] = s;
  }
  __syncthreads();
  const float scale = 0.08838834764831845f;  // 1/sqrt(128)
  for (int idx = tid; idx < M_ * M_; idx += 256) {
    int q = idx >> 5, k2 = idx & 31;
    float s = 0.f;
    for (int dd = 0; dd < D_; ++dd) s += qk[q * 384 + dd] * qk[k2 * 384 + 128 + dd];
    dm[idx] = s * scale;
  }
  __syncthreads();
  if (tid < M_) {
    int q = tid;
    float mx = -1e30f;
    for (int k2 = 0; k2 < M_; ++k2) mx = fmaxf(mx, dm[q * M_ + k2]);
    float sm = 0.f;
    for (int k2 = 0; k2 < M_; ++k2) { float e = __expf(dm[q * M_ + k2] - mx); aw[q * M_ + k2] = e; sm += e; }
    float inv = 1.f / sm;
    for (int k2 = 0; k2 < M_; ++k2) {
      float w = aw[q * M_ + k2] * inv;
      aw[q * M_ + k2] = w;
      attn_out[(size_t)bh * M_ * M_ + q * M_ + k2] = w;
    }
  }
  __syncthreads();
  for (int idx = tid; idx < M_ * D_; idx += 256) {
    int q = idx >> 7, dd = idx & 127;
    float s = 0.f;
    for (int k2 = 0; k2 < M_; ++k2) s += aw[q * M_ + k2] * qk[k2 * 384 + 256 + dd];
    otok[(size_t)bh * M_ * D_ + idx] = s;
  }
}

// ------------------------------------------------- scatter back to tokens
__global__ __launch_bounds__(256) void k_scatter(
    const float* __restrict__ otok, const float* __restrict__ swp,
    __bf16* __restrict__ hidden) {
  __shared__ float ot[M_ * D_];
  __shared__ float wc[M_ * 64];
  const int tid = threadIdx.x;
  const long tg = (long)blockIdx.x * 64;
  const int b = (int)(tg / N_), n0 = (int)(tg % N_);

  for (int h = 0; h < H_; ++h) {
    int bh = b * H_ + h;
    for (int idx = tid; idx < M_ * D_; idx += 256) ot[idx] = otok[(size_t)bh * M_ * D_ + idx];
    for (int idx = tid; idx < M_ * 64; idx += 256) {
      int m = idx >> 6, t2 = idx & 63;
      wc[idx] = swp[((size_t)(bh * M_ + m)) * N_ + n0 + t2];
    }
    __syncthreads();
    for (int idx = tid; idx < 64 * D_; idx += 256) {
      int tok = idx >> 7, dd = idx & 127;
      float s = 0.f;
#pragma unroll
      for (int m = 0; m < M_; ++m) s += wc[m * 64 + tok] * ot[m * D_ + dd];
      hidden[((size_t)(b * N_ + n0 + tok)) * HD_ + h * D_ + dd] = (__bf16)s;
    }
    __syncthreads();
  }
}

// ---------------------------------------------------------------- launcher
extern "C" void kernel_launch(void* const* d_in, const int* in_sizes, int n_in,
                              void* d_out, int out_size, void* d_ws, size_t ws_size,
                              hipStream_t stream) {
  (void)in_sizes; (void)n_in; (void)out_size; (void)ws_size;
  const float* x      = (const float*)d_in[0];
  const float* wkv    = (const float*)d_in[1];
  const float* kvbias = (const float*)d_in[2];
  const float* wtq    = (const float*)d_in[3];
  const float* wtqb   = (const float*)d_in[4];
  const float* qkvw   = (const float*)d_in[5];
  const float* wout   = (const float*)d_in[6];
  const float* outb   = (const float*)d_in[7];
  float* out = (float*)d_out;

  char* ws = (char*)d_ws;
  size_t off = 0;
  auto alloc = [&](size_t bytes) -> char* {
    char* p = ws + off;
    off += (bytes + 255) & ~(size_t)255;
    return p;
  };
  __bf16* xb     = (__bf16*)alloc((size_t)BN_TOT * HD_ * 2);
  __bf16* wkvb   = (__bf16*)alloc((size_t)2 * HD_ * HD_ * 2);
  __bf16* woutb  = (__bf16*)alloc((size_t)HD_ * HD_ * 2);
  __bf16* kvb    = (__bf16*)alloc((size_t)BN_TOT * 2 * HD_ * 2);
  __bf16* hidden = (__bf16*)alloc((size_t)BN_TOT * HD_ * 2);
  float*  stok   = (float*)alloc((size_t)B_ * H_ * M_ * D_ * 4);
  float*  snorm  = (float*)alloc((size_t)B_ * H_ * M_ * 4);
  float*  otok   = (float*)alloc((size_t)B_ * H_ * M_ * D_ * 4);

  float* swp  = out + (size_t)BN_TOT * HD_;  // slice_weights region
  float* attn = out + 83886337;              // attn_weights region

  k_f32_to_bf16<<<4096, 256, 0, stream>>>(x,    xb,    BN_TOT * (long)HD_);
  k_f32_to_bf16<<<2048, 256, 0, stream>>>(wkv,  wkvb,  (long)2 * HD_ * HD_);
  k_f32_to_bf16<<<1024, 256, 0, stream>>>(wout, woutb, (long)HD_ * HD_);
  k_zero_f32<<<512, 256, 0, stream>>>(stok,  (long)B_ * H_ * M_ * D_);
  k_zero_f32<<<4,   256, 0, stream>>>(snorm, (long)B_ * H_ * M_);

  dim3 g1(2 * HD_ / BNT, BN_TOT / BM);  // (16, 512)
  gemm_bf16<true><<<g1, 256, 0, stream>>>(xb, wkvb, kvbias, (void*)kvb,
                                          (int)BN_TOT, 2 * HD_, HD_);

  k_scores<<<BN_TOT / 128, 256, 0, stream>>>(kvb, wtq, swp, stok, snorm);
  k_token_attn<<<B_ * H_, 256, 0, stream>>>(stok, snorm, qkvw, attn, otok);
  k_scatter<<<BN_TOT / 64, 256, 0, stream>>>(otok, swp, hidden);

  dim3 g2(HD_ / BNT, BN_TOT / BM);  // (8, 512)
  gemm_bf16<false><<<g2, 256, 0, stream>>>(hidden, woutb, outb, (void*)out,
                                           (int)BN_TOT, HD_, HD_);

  k_misc<<<1, 256, 0, stream>>>(wtqb, out);
}
